// GptOssAttention_13374528160270
// MI455X (gfx1250) — compile-verified, matrix operand order checked by reading, not run on hardware
//
#include <hip/hip_runtime.h>
#include <math.h>

// ---------------- problem constants ----------------
#define S_LEN   1536
#define DMODEL  2880
#define HQ      64
#define HKV     8
#define DH      64
#define QDIM    (HQ*DH)     // 4096
#define KVDIM   (HKV*DH)    // 512
#define SCALE   0.125f
#define NEGBIG  (-1.0e30f)

typedef __attribute__((ext_vector_type(16))) _Float16 v16h;
typedef __attribute__((ext_vector_type(8)))  float    v8f;

// ---- CDNA5 async global->LDS copy ----
// The clang builtin exists but wants LangAS-qualified int4 pointers; use the
// documented inline-asm path instead (portable across toolchains). Feature-gate
// on the builtin's presence: builtin present => MC knows the instruction.
#if __has_builtin(__builtin_amdgcn_global_load_async_to_lds_b128)
  #define HAVE_ASYNC_LDS 1
  #if __has_builtin(__builtin_amdgcn_s_wait_asynccnt)
    #define WAIT_ASYNC(n) __builtin_amdgcn_s_wait_asynccnt(n)
  #else
    #define WAIT_ASYNC(n) asm volatile("s_wait_asynccnt %0" :: "n"(n) : "memory")
  #endif
  __device__ __forceinline__ void async_copy_b128(const _Float16* g, const _Float16* l) {
      // generic LDS address low 32 bits = LDS offset
      unsigned lds = (unsigned)(uintptr_t)l;
      asm volatile("global_load_async_to_lds_b128 %0, %1, off"
                   :: "v"(lds), "v"(g) : "memory");
  }
#else
  #define HAVE_ASYNC_LDS 0
#endif

// ---------------- WMMA fragment loaders (CDNA5 wave32 layouts) ----------------
// A-matrix 16x32 f16: lane m = lane&15; VGPR j holds halves (2j,2j+1);
// k(j) = 16*(j/4) + 8*(lane>=16) + 2*(j%4); contiguous k pairs.
__device__ __forceinline__ v16h load_frag_rowk(const _Float16* __restrict__ base, int ld, int lane) {
    v16h a;
    const int m  = lane & 15;
    const int hi = (lane >> 4) << 3;            // 0 or 8
    const _Float16* row = base + (size_t)m * ld;
#pragma unroll
    for (int j = 0; j < 8; ++j) {
        int k = ((j >> 2) << 4) + hi + ((j & 3) << 1);
        a[2*j]   = row[k];
        a[2*j+1] = row[k+1];
    }
    return a;
}

// B-matrix 32x16 f16 (K along rows of src): lane n = lane&15, same K packing.
__device__ __forceinline__ v16h load_frag_colk(const _Float16* __restrict__ base, int ld, int lane) {
    v16h b;
    const int n  = lane & 15;
    const int hi = (lane >> 4) << 3;
#pragma unroll
    for (int j = 0; j < 8; ++j) {
        int k = ((j >> 2) << 4) + hi + ((j & 3) << 1);
        b[2*j]   = base[(size_t)k      * ld + n];
        b[2*j+1] = base[(size_t)(k+1)  * ld + n];
    }
    return b;
}

// A-fragment sourced from fp32 LDS probabilities, converted to f16 on the fly.
__device__ __forceinline__ v16h load_frag_probs(const float* __restrict__ probs, int ld, int kb, int lane) {
    v16h a;
    const int m  = lane & 15;
    const int hi = (lane >> 4) << 3;
    const float* row = probs + (size_t)m * ld + kb;
#pragma unroll
    for (int j = 0; j < 8; ++j) {
        int k = ((j >> 2) << 4) + hi + ((j & 3) << 1);
        a[2*j]   = (_Float16)row[k];
        a[2*j+1] = (_Float16)row[k+1];
    }
    return a;
}

#define WMMA_F16(a,b,c) __builtin_amdgcn_wmma_f32_16x16x32_f16(false,(a),false,(b),(short)0,(c),false,false)

// ---------------- f32 -> f16 convert ----------------
__global__ __launch_bounds__(256) void cvt_f32_f16(const float* __restrict__ src,
                                                   _Float16* __restrict__ dst, int n) {
    int i = blockIdx.x * 256 + threadIdx.x;
    if (i < n) dst[i] = (_Float16)src[i];
}

// ---------------- generic GEMM: Y[M,N] = X[M,K] * W[N,K]^T + bias ----------------
// block = 256 threads = 8 waves; block tile 32(M) x 128(N); wave tile 16x32.
// X tile staged through LDS; on CDNA5 the staging uses double-buffered
// GLOBAL_LOAD_ASYNC_TO_LDS_B128 (ASYNCcnt) overlapped with WMMA compute.
__global__ __launch_bounds__(256)
void gemm_xwT_bias(const _Float16* __restrict__ X, const _Float16* __restrict__ W,
                   const float* __restrict__ bias, float* __restrict__ Y,
                   int M, int N, int K) {
    const int tid  = threadIdx.x;
    const int wave = tid >> 5, lane = tid & 31;
    const int m0   = blockIdx.y * 32;
    const int mloc = (wave >> 2) << 4;                 // 0 / 16
    const int ncol = blockIdx.x * 128 + ((wave & 3) << 5);
    const bool active = (ncol < N);
    v8f c0 = {}; v8f c1 = {};

#if HAVE_ASYNC_LDS
    // row stride 40 halfs = 80 B (multiple of 16 B for b128 chunks)
    __shared__ _Float16 xs[2][32][40];
    const int nk = K / 32;
    // async issue: threads 0..127 each move one 16-byte chunk of the 32x32 tile
    const int arow = tid >> 2;
    const int ac16 = (tid & 3) << 3;
    const _Float16* agbase = X + (size_t)(m0 + arow) * K + ac16;

    if (tid < 128) async_copy_b128(agbase, &xs[0][arow][ac16]);
    for (int i = 0; i < nk; ++i) {
        const int kb = i * 32;
        if (i + 1 < nk) {
            if (tid < 128) async_copy_b128(agbase + kb + 32, &xs[(i + 1) & 1][arow][ac16]);
            WAIT_ASYNC(1);                 // stage i complete, stage i+1 in flight
        } else {
            WAIT_ASYNC(0);
        }
        __syncthreads();
        if (active) {
            const _Float16* wp0 = W + (size_t)ncol * K + kb;
            const _Float16* wp1 = W + (size_t)(ncol + 16) * K + kb;
            if (kb + 32 < K) {             // prefetch next W fragments (global_prefetch_b8)
                __builtin_prefetch((const void*)(wp0 + 32), 0, 3);
                __builtin_prefetch((const void*)(wp1 + 32), 0, 3);
            }
            v16h a  = load_frag_rowk(&xs[i & 1][mloc][0], 40, lane);
            v16h b0 = load_frag_rowk(wp0, K, lane);
            v16h b1 = load_frag_rowk(wp1, K, lane);
            c0 = WMMA_F16(a, b0, c0);
            c1 = WMMA_F16(a, b1, c1);
        }
        __syncthreads();                   // tile being read is re-issued only after this
    }
#else
    __shared__ _Float16 xs[32][34];
    for (int kb = 0; kb < K; kb += 32) {
        const int row  = tid >> 3;
        const int col4 = (tid & 7) << 2;
        const _Float16* src = X + (size_t)(m0 + row) * K + kb + col4;
#pragma unroll
        for (int t = 0; t < 4; ++t) xs[row][col4 + t] = src[t];
        __syncthreads();
        if (active) {
            v16h a  = load_frag_rowk(&xs[mloc][0], 34, lane);
            v16h b0 = load_frag_rowk(W + (size_t)ncol * K + kb, K, lane);
            v16h b1 = load_frag_rowk(W + (size_t)(ncol + 16) * K + kb, K, lane);
            c0 = WMMA_F16(a, b0, c0);
            c1 = WMMA_F16(a, b1, c1);
        }
        __syncthreads();
    }
#endif

    if (active) {
        const int n  = lane & 15;
        const int mo = (lane >> 4) << 3;
#pragma unroll
        for (int r = 0; r < 8; ++r) {
            const size_t m = (size_t)(m0 + mloc + mo + r);
            Y[m * N + ncol      + n] = c0[r] + bias[ncol      + n];
            Y[m * N + ncol + 16 + n] = c1[r] + bias[ncol + 16 + n];
        }
    }
}

// ---------------- RoPE + repack [S][nh*64] f32 -> [nh][S][64] f16 ----------------
__global__ __launch_bounds__(256)
void rope_pack(const float* __restrict__ src, _Float16* __restrict__ dst,
               const float* __restrict__ cosb, const float* __restrict__ sinb,
               int nh, int do_rope) {
    int idx = blockIdx.x * 256 + threadIdx.x;
    if (do_rope) {
        if (idx >= nh * S_LEN * 32) return;
        const int d = idx & 31;
        const int s = (idx >> 5) % S_LEN;
        const int h = idx / (S_LEN * 32);
        const float c  = cosb[s * 32 + d];
        const float sn = sinb[s * 32 + d];
        const float* row = src + (size_t)s * nh * 64 + h * 64;
        const float x1 = row[d], x2 = row[d + 32];
        _Float16* drow = dst + ((size_t)h * S_LEN + s) * 64;
        drow[d]      = (_Float16)(x1 * c - x2 * sn);
        drow[d + 32] = (_Float16)(x2 * c + x1 * sn);
    } else {
        if (idx >= nh * S_LEN * 64) return;
        const int d = idx & 63;
        const int s = (idx >> 6) % S_LEN;
        const int h = idx / (S_LEN * 64);
        dst[((size_t)h * S_LEN + s) * 64 + d] =
            (_Float16)src[(size_t)s * nh * 64 + h * 64 + d];
    }
}

// ---------------- fused attention: scores + sink-softmax + attn out + PV ----------------
// grid = (S/16, HQ); 128 threads (4 waves); dynamic LDS:
//   probs[16][1536] f32 | rowI[16(+16)] | po[4][16][64]
__global__ __launch_bounds__(128)
void attention_kernel(const _Float16* __restrict__ qh, const _Float16* __restrict__ kh,
                      const _Float16* __restrict__ vh, const float* __restrict__ sinks,
                      float* __restrict__ attn_out, _Float16* __restrict__ ao) {
    extern __shared__ char smem[];
    float* probs = (float*)smem;                 // 16*1536
    float* rowI  = probs + 16 * S_LEN;           // 16 (+16 pad)
    float* po    = rowI + 32;                    // 4*16*64

    const int qt = blockIdx.x, h = blockIdx.y;
    const int qbase = qt * 16, qend = qbase + 15;
    const int kv = h >> 3;                       // N_REP = 8
    const int tid = threadIdx.x, wave = tid >> 5, lane = tid & 31;
    const int n  = lane & 15;
    const int mo = (lane >> 4) << 3;

    // Q fragments for this 16-row tile (d = 0..31 and 32..63)
    const _Float16* qp = qh + ((size_t)h * S_LEN + qbase) * 64;
    const v16h aq0 = load_frag_rowk(qp,      64, lane);
    const v16h aq1 = load_frag_rowk(qp + 32, 64, lane);

    // ---- scores: waves split the 96 key tiles ----
    for (int j = wave; j < S_LEN / 16; j += 4) {
        const int kb = j * 16;
        if (kb > qend) {                         // fully masked (future) tile
#pragma unroll
            for (int r = 0; r < 8; ++r) probs[(size_t)(mo + r) * S_LEN + kb + n] = NEGBIG;
            continue;
        }
        const _Float16* kp = kh + ((size_t)kv * S_LEN + kb) * 64;
        v16h b0 = load_frag_rowk(kp,      64, lane);
        v16h b1 = load_frag_rowk(kp + 32, 64, lane);
        v8f c = {};
        c = WMMA_F16(aq0, b0, c);
        c = WMMA_F16(aq1, b1, c);
#pragma unroll
        for (int r = 0; r < 8; ++r) {
            const int m = mo + r;
            float sv = c[r] * SCALE;
            if (kb + n > qbase + m) sv = NEGBIG; // causal
            probs[(size_t)m * S_LEN + kb + n] = sv;
        }
    }
    __syncthreads();

    // ---- softmax with sink column: 8 threads per row ----
    {
        const int r = tid >> 3, sub = tid & 7;
        const float sinkv = sinks[h];
        float lm = -3.4e38f;
        for (int k = sub; k < S_LEN; k += 8) lm = fmaxf(lm, probs[(size_t)r * S_LEN + k]);
#pragma unroll
        for (int off = 4; off; off >>= 1) lm = fmaxf(lm, __shfl_xor(lm, off, 32));
        const float Mx = fmaxf(lm, sinkv);
        float ls = 0.f;
        for (int k = sub; k < S_LEN; k += 8) {
            const float e = expf(probs[(size_t)r * S_LEN + k] - Mx);
            probs[(size_t)r * S_LEN + k] = e;
            ls += e;
        }
#pragma unroll
        for (int off = 4; off; off >>= 1) ls += __shfl_xor(ls, off, 32);
        if (sub == 0) rowI[r] = 1.0f / (ls + expf(sinkv - Mx));
    }
    __syncthreads();

    // ---- normalize, write attn rows to global (coalesced) ----
    for (int r = 0; r < 16; ++r) {
        const float inv = rowI[r];
        const size_t gbase = ((size_t)h * S_LEN + (qbase + r)) * S_LEN;
        for (int k = tid; k < S_LEN; k += 128) {
            const float p = probs[(size_t)r * S_LEN + k] * inv;
            probs[(size_t)r * S_LEN + k] = p;
            attn_out[gbase + k] = p;
        }
    }
    __syncthreads();

    // ---- PV: waves split the 48 K-chunks of 32 ----
    v8f o0 = {}, o1 = {}, o2 = {}, o3 = {};
    for (int ch = wave; ch < S_LEN / 32; ch += 4) {
        const int kb = ch * 32;
        if (kb > qend) continue;                 // probs are exactly zero there
        const v16h ap = load_frag_probs(probs, S_LEN, kb, lane);
        const _Float16* vp = vh + ((size_t)kv * S_LEN + kb) * 64;
        o0 = WMMA_F16(ap, load_frag_colk(vp,      64, lane), o0);
        o1 = WMMA_F16(ap, load_frag_colk(vp + 16, 64, lane), o1);
        o2 = WMMA_F16(ap, load_frag_colk(vp + 32, 64, lane), o2);
        o3 = WMMA_F16(ap, load_frag_colk(vp + 48, 64, lane), o3);
    }
    {   // partials to LDS
        float* pw = po + (size_t)wave * 16 * 64;
#pragma unroll
        for (int r = 0; r < 8; ++r) {
            const int m = mo + r;
            pw[m * 64      + n] = o0[r];
            pw[m * 64 + 16 + n] = o1[r];
            pw[m * 64 + 32 + n] = o2[r];
            pw[m * 64 + 48 + n] = o3[r];
        }
    }
    __syncthreads();
    for (int idx = tid; idx < 16 * 64; idx += 128) {
        const float s = po[idx] + po[1024 + idx] + po[2048 + idx] + po[3072 + idx];
        const int m = idx >> 6, d = idx & 63;
        ao[(size_t)(qbase + m) * QDIM + h * 64 + d] = (_Float16)s;
    }
}

// ---------------- host side ----------------
extern "C" void kernel_launch(void* const* d_in, const int* in_sizes, int n_in,
                              void* d_out, int out_size, void* d_ws, size_t ws_size,
                              hipStream_t stream) {
    const float* hs    = (const float*)d_in[0];
    const float* cosb  = (const float*)d_in[2];
    const float* sinb  = (const float*)d_in[3];
    const float* Wq    = (const float*)d_in[4];
    const float* bq    = (const float*)d_in[5];
    const float* Wk    = (const float*)d_in[6];
    const float* bk    = (const float*)d_in[7];
    const float* Wv    = (const float*)d_in[8];
    const float* bv    = (const float*)d_in[9];
    const float* Wo    = (const float*)d_in[10];
    const float* bo    = (const float*)d_in[11];
    const float* sinks = (const float*)d_in[12];

    float* out  = (float*)d_out;                       // [S, DM]
    float* attn = out + (size_t)S_LEN * DMODEL;        // [HQ, S, S]

    // workspace carve
    char*  ws  = (char*)d_ws;
    size_t off = 0;
    auto carve = [&](size_t bytes) -> void* {
        void* p = ws + off;
        off = (off + bytes + 255) & ~(size_t)255;
        return p;
    };
    _Float16* hsF = (_Float16*)carve((size_t)S_LEN * DMODEL * 2);
    _Float16* WqF = (_Float16*)carve((size_t)QDIM  * DMODEL * 2);
    _Float16* WkF = (_Float16*)carve((size_t)KVDIM * DMODEL * 2);
    _Float16* WvF = (_Float16*)carve((size_t)KVDIM * DMODEL * 2);
    _Float16* WoF = (_Float16*)carve((size_t)DMODEL * QDIM * 2);
    float*    Qf  = (float*)carve((size_t)S_LEN * QDIM  * 4);
    float*    Kf  = (float*)carve((size_t)S_LEN * KVDIM * 4);
    float*    Vf  = (float*)carve((size_t)S_LEN * KVDIM * 4);
    _Float16* qhB = (_Float16*)carve((size_t)HQ  * S_LEN * DH * 2);
    _Float16* khB = (_Float16*)carve((size_t)HKV * S_LEN * DH * 2);
    _Float16* vhB = (_Float16*)carve((size_t)HKV * S_LEN * DH * 2);
    _Float16* ao  = WqF;   // alias: Wq f16 dead after Q projection

    const int att_lds = (16 * S_LEN + 32 + 4 * 16 * 64) * (int)sizeof(float);
    (void)hipFuncSetAttribute((const void*)attention_kernel,
                              hipFuncAttributeMaxDynamicSharedMemorySize, att_lds);

    auto cvt = [&](const float* s, _Float16* d, size_t n) {
        cvt_f32_f16<<<dim3((unsigned)((n + 255) / 256)), 256, 0, stream>>>(s, d, (int)n);
    };
    cvt(hs, hsF, (size_t)S_LEN * DMODEL);
    cvt(Wq, WqF, (size_t)QDIM  * DMODEL);
    cvt(Wk, WkF, (size_t)KVDIM * DMODEL);
    cvt(Wv, WvF, (size_t)KVDIM * DMODEL);
    cvt(Wo, WoF, (size_t)DMODEL * QDIM);

    // QKV projections
    gemm_xwT_bias<<<dim3(QDIM / 128, S_LEN / 32), 256, 0, stream>>>(
        hsF, WqF, bq, Qf, S_LEN, QDIM, DMODEL);
    gemm_xwT_bias<<<dim3(KVDIM / 128, S_LEN / 32), 256, 0, stream>>>(
        hsF, WkF, bk, Kf, S_LEN, KVDIM, DMODEL);
    gemm_xwT_bias<<<dim3(KVDIM / 128, S_LEN / 32), 256, 0, stream>>>(
        hsF, WvF, bv, Vf, S_LEN, KVDIM, DMODEL);

    // RoPE + head-major f16 repack
    rope_pack<<<dim3((HQ  * S_LEN * 32 + 255) / 256), 256, 0, stream>>>(Qf, qhB, cosb, sinb, HQ,  1);
    rope_pack<<<dim3((HKV * S_LEN * 32 + 255) / 256), 256, 0, stream>>>(Kf, khB, cosb, sinb, HKV, 1);
    rope_pack<<<dim3((HKV * S_LEN * 64 + 255) / 256), 256, 0, stream>>>(Vf, vhB, cosb, sinb, HKV, 0);

    // fused attention (writes attn probs to d_out and f16 context to ao)
    attention_kernel<<<dim3(S_LEN / 16, HQ), 128, att_lds, stream>>>(
        qhB, khB, vhB, sinks, attn, ao);

    // output projection
    gemm_xwT_bias<<<dim3((DMODEL + 127) / 128, S_LEN / 32), 256, 0, stream>>>(
        ao, WoF, bo, out, S_LEN, DMODEL, QDIM);
}
